// Multi_Heads_Self_Attn_1812476199807
// MI455X (gfx1250) — matrix operand order repaired and checked by reading, not compiled, not run on hardware
//
#include <hip/hip_runtime.h>

// ---------------- problem constants ----------------
#define BSZ   8
#define CCH   64
#define NTOK  1728          // 12*12*12 spatial tokens
#define HEADS 4
#define DKV   64
#define HKV   256
#define NT16  108           // NTOK/16
#define NPAIR 54            // NTOK/32
#define EPSV  1e-5f
#define LEAKV 0.01f

#define OUT_ELEMS   884736u           // B*C*N
#define W_ELEMS     95551488u         // B*H*N*N
#define FRAG_PER_BH 216               // 108 tiles * 2 kchunks  (== 54*4 for V)
#define EL_PER_BH   110592            // FRAG_PER_BH * 512

typedef __bf16 v16bf __attribute__((ext_vector_type(16)));
typedef float  v8f   __attribute__((ext_vector_type(8)));
typedef float  v4f   __attribute__((ext_vector_type(4)));

__device__ __forceinline__ unsigned short f2bfu(float f) {
  union { float f; unsigned u; } a; a.f = f;
  unsigned u = a.u + 0x7FFFu + ((a.u >> 16) & 1u);   // round-to-nearest-even
  return (unsigned short)(u >> 16);
}

// ------------------------------------------------------------------
// Stage Q/K/V as bf16 WMMA fragments (CDNA5 ISA VGPR layouts).
//   mode 0: Q  -> A-operand 16x32 layout (K = e + (e>=8?8:0) + (lane>=16?8:0))
//   mode 1: K  -> B-operand 32x16 layout of K^T (K = e + (e>=8?8:0) + (lane>=16?8:0))
//   mode 2: V  -> B-operand 32x16 layout, rows = tokens, cols = dk
// Fragment element order is exactly (bh, fragId, lane, e) so dst[i] = value.
// ------------------------------------------------------------------
__global__ void __launch_bounds__(256)
proj_frag_kernel(const float* __restrict__ x, const float* __restrict__ w,
                 unsigned short* __restrict__ dst, int mode) {
  long i = (long)blockIdx.x * blockDim.x + threadIdx.x;
  int bh     = (int)(i / EL_PER_BH);
  int rem    = (int)(i % EL_PER_BH);
  int b      = bh >> 2, h = bh & 3;
  int fragId = rem >> 9;
  int within = rem & 511;
  int lane   = within >> 4;
  int e      = within & 15;
  int tok, o;
  if (mode == 0) {
    int mTile = fragId >> 1, kc = fragId & 1;
    tok = mTile * 16 + (lane & 15);
    int kred = kc * 32 + e + ((e >= 8) ? 8 : 0) + ((lane >= 16) ? 8 : 0);
    o = h * DKV + kred;
  } else if (mode == 1) {
    int cTile = fragId >> 1, kc = fragId & 1;
    tok = cTile * 16 + (lane & 15);
    int kred = kc * 32 + e + ((e >= 8) ? 8 : 0) + ((lane >= 16) ? 8 : 0);
    o = h * DKV + kred;
  } else {
    int mChunk = fragId >> 2, dkTile = fragId & 3;
    tok = mChunk * 32 + e + ((e >= 8) ? 8 : 0) + ((lane >= 16) ? 8 : 0);
    o = h * DKV + dkTile * 16 + (lane & 15);
  }
  const float* xb = x + (long)b * CCH * NTOK + tok;
  const float* wr = w + o * CCH;
  float acc = 0.f;
  #pragma unroll
  for (int c = 0; c < CCH; ++c) acc += xb[(long)c * NTOK] * wr[c];
  dst[i] = f2bfu(acc);
}

// ------------------------------------------------------------------
// Fused attention: S = (Q K^T)/N  -> NT-store weights (f32) -> P.V accum.
// grid = B*H*NT16 blocks, 128 threads (4 waves). bf16 WMMA, f32 accum.
// weights is written once, never re-read -> non-temporal stores keep the
// 14 MB of K/V fragments resident in the 192 MB L2 across all 3456 blocks.
// ------------------------------------------------------------------
__global__ void __launch_bounds__(128)
attn_kernel(const unsigned short* __restrict__ qfrag,
            const unsigned short* __restrict__ kfrag,
            const unsigned short* __restrict__ vfrag,
            float* __restrict__ weightsOut, float* __restrict__ Obuf) {
  __shared__ __align__(16) unsigned short pstage[4][16 * 32]; // per-wave P tile
  __shared__ float ldsO[16 * 64];                             // O accumulator

  int blk   = blockIdx.x;
  int mTile = blk % NT16;
  int bh    = blk / NT16;
  int b     = bh >> 2, h = bh & 3;
  int tid   = threadIdx.x;
  int wave  = tid >> 5;
  int lane  = tid & 31;
  int lo    = lane & 15;
  int hiA   = (lane >> 4) ? 8 : 0;      // A/C layout row/K offset for upper half-wave
  int sel   = (lane >> 4) & 1;

  for (int idx = tid; idx < 16 * 64; idx += 128) ldsO[idx] = 0.f;
  __syncthreads();

  const unsigned short* qb = qfrag + ((long)bh * FRAG_PER_BH + mTile * 2) * 512 + lane * 16;
  v16bf qa0 = *(const v16bf*)(qb);
  v16bf qa1 = *(const v16bf*)(qb + 512);

  v8f o0 = {}, o1 = {}, o2 = {}, o3 = {};
  const float invN = 1.0f / (float)NTOK;

  unsigned short* ps = &pstage[wave][0];
  const v4f* psv = (const v4f*)ps;

  for (int pair = wave; pair < NPAIR; pair += 4) {
    // ---- S = (Q K^T)/N for two adjacent 16x16 column tiles (one 16x32 P) ----
    const unsigned short* kb =
        kfrag + ((long)bh * FRAG_PER_BH + pair * 4) * 512 + lane * 16;
    __builtin_prefetch(kb + 4 * 2048, 0, 1);          // next pair's K frags
    v16bf k00 = *(const v16bf*)(kb);
    v16bf k01 = *(const v16bf*)(kb + 512);
    v16bf k10 = *(const v16bf*)(kb + 1024);
    v16bf k11 = *(const v16bf*)(kb + 1536);
    v8f c0 = {}, c1 = {};
    c0 = __builtin_amdgcn_wmma_f32_16x16x32_bf16(false, qa0, false, k00, (short)0, c0, false, false);
    c0 = __builtin_amdgcn_wmma_f32_16x16x32_bf16(false, qa1, false, k01, (short)0, c0, false, false);
    c1 = __builtin_amdgcn_wmma_f32_16x16x32_bf16(false, qa0, false, k10, (short)0, c1, false, false);
    c1 = __builtin_amdgcn_wmma_f32_16x16x32_bf16(false, qa1, false, k11, (short)0, c1, false, false);
    // scale -> streaming store to weights + stage bf16 P tile in LDS
    size_t wbase = ((size_t)bh * NTOK + (size_t)(mTile * 16 + hiA)) * NTOK + pair * 32 + lo;
    #pragma unroll
    for (int r = 0; r < 8; ++r) {
      float w0 = c0[r] * invN;
      float w1 = c1[r] * invN;
      __builtin_nontemporal_store(w0, &weightsOut[wbase + (size_t)r * NTOK]);
      __builtin_nontemporal_store(w1, &weightsOut[wbase + (size_t)r * NTOK + 16]);
      ps[(r + hiA) * 32 +      lo] = f2bfu(w0);
      ps[(r + hiA) * 32 + 16 + lo] = f2bfu(w1);
    }
    asm volatile("" ::: "memory");   // DS in-order per wave; stop compiler reordering
    // reload P in A-operand 16x32 layout: two ds_load_b128 per lane
    union { v4f q[2]; v16bf v; } pu;
    pu.q[0] = psv[lo * 4 + sel];
    pu.q[1] = psv[lo * 4 + 2 + sel];
    v16bf pa = pu.v;
    asm volatile("" ::: "memory");
    // ---- O += P . V over this 32-token chunk, 4 dk tiles ----
    const unsigned short* vb0 =
        vfrag + ((long)bh * FRAG_PER_BH + pair * 4) * 512 + lane * 16;
    __builtin_prefetch(vb0 + 4 * 2048, 0, 1);         // next pair's V frags
    v16bf vb;
    vb = *(const v16bf*)(vb0);
    o0 = __builtin_amdgcn_wmma_f32_16x16x32_bf16(false, pa, false, vb, (short)0, o0, false, false);
    vb = *(const v16bf*)(vb0 + 512);
    o1 = __builtin_amdgcn_wmma_f32_16x16x32_bf16(false, pa, false, vb, (short)0, o1, false, false);
    vb = *(const v16bf*)(vb0 + 1024);
    o2 = __builtin_amdgcn_wmma_f32_16x16x32_bf16(false, pa, false, vb, (short)0, o2, false, false);
    vb = *(const v16bf*)(vb0 + 1536);
    o3 = __builtin_amdgcn_wmma_f32_16x16x32_bf16(false, pa, false, vb, (short)0, o3, false, false);
  }
  // combine the 4 waves' partial P.V sums via LDS float atomics (ds_add_f32)
  #pragma unroll
  for (int r = 0; r < 8; ++r) {
    int row = r + hiA;
    atomicAdd(&ldsO[row * 64 +      lo], o0[r]);
    atomicAdd(&ldsO[row * 64 + 16 + lo], o1[r]);
    atomicAdd(&ldsO[row * 64 + 32 + lo], o2[r]);
    atomicAdd(&ldsO[row * 64 + 48 + lo], o3[r]);
  }
  __syncthreads();
  // write merged-head O[b, token, h*64+dk] (f32), 8 contiguous floats/thread
  int row  = tid >> 3;
  int colb = (tid & 7) * 8;
  float* dstp = Obuf + ((long)b * NTOK + mTile * 16 + row) * HKV + h * DKV + colb;
  const v4f* sv = (const v4f*)&ldsO[row * 64 + colb];
  ((v4f*)dstp)[0] = sv[0];
  ((v4f*)dstp)[1] = sv[1];
}

// ------------------------------------------------------------------
// Output 1x1x1 conv + per-(b,c) sum / sumsq for InstanceNorm.
// O rows staged into LDS with GLOBAL_LOAD_ASYNC_TO_LDS_B128 (ASYNCcnt).
// grid = B * 432 blocks (4 tokens each), 256 threads.
// ------------------------------------------------------------------
__global__ void __launch_bounds__(256)
outconv_kernel(const float* __restrict__ Obuf, const float* __restrict__ wo,
               float* __restrict__ Ybuf, float* __restrict__ gsum,
               float* __restrict__ gsumsq) {
  __shared__ __align__(16) float xs[4][HKV];
  int blk = blockIdx.x;
  int b   = blk / 432;
  int n0  = (blk % 432) * 4;
  int tid = threadIdx.x;
  {
    int i  = tid >> 6;
    int c4 = tid & 63;
    const float* gp = Obuf + ((long)b * NTOK + n0 + i) * HKV + c4 * 4;
    // low 32 bits of a flat shared pointer == wave-relative LDS byte offset
    unsigned ldsoff = (unsigned)(unsigned long long)(&xs[i][c4 * 4]);
    unsigned long long ga = (unsigned long long)gp;
    asm volatile("global_load_async_to_lds_b128 %0, %1, off"
                 :: "v"(ldsoff), "v"(ga) : "memory");
  }
  asm volatile("s_wait_asynccnt 0x0" ::: "memory");
  __syncthreads();
  int o = tid & 63;
  int i = tid >> 6;
  const float* wr = wo + o * HKV;
  float acc = 0.f;
  #pragma unroll 8
  for (int c = 0; c < HKV; ++c) acc += wr[c] * xs[i][c];
  Ybuf[((long)b * CCH + o) * NTOK + n0 + i] = acc;
  atomicAdd(&gsum[b * CCH + o], acc);          // global_atomic_add_f32
  atomicAdd(&gsumsq[b * CCH + o], acc * acc);
}

__global__ void zero_stats_kernel(float* __restrict__ g) {
  for (int i = threadIdx.x; i < 1024; i += 256) g[i] = 0.f;
}

// ------------------------------------------------------------------
// InstanceNorm3d (affine=False) + residual + LeakyReLU, plus gamma copy.
// Ybuf is last-use (NT load); out is never re-read (NT store).
// ------------------------------------------------------------------
__global__ void __launch_bounds__(256)
norm_kernel(const float* __restrict__ Ybuf, const float* __restrict__ x,
            const float* __restrict__ gsum, const float* __restrict__ gsumsq,
            const float* __restrict__ gamma_in, float* __restrict__ dout) {
  int bc = blockIdx.x;                       // b*64 + c
  float mean = gsum[bc] * (1.0f / NTOK);
  float var  = gsumsq[bc] * (1.0f / NTOK) - mean * mean;
  float rs   = rsqrtf(var + EPSV);
  const float* y  = Ybuf + (long)bc * NTOK;
  const float* xr = x + (long)bc * NTOK;
  float* od = dout + (long)bc * NTOK;
  for (int n = threadIdx.x; n < NTOK; n += 256) {
    float yv = __builtin_nontemporal_load(&y[n]);
    float v  = (yv - mean) * rs + xr[n];
    v = (v >= 0.f) ? v : LEAKV * v;
    __builtin_nontemporal_store(v, &od[n]);
  }
  if (bc == 0 && threadIdx.x == 0)
    dout[(size_t)OUT_ELEMS + W_ELEMS] = gamma_in[0];   // gamma passthrough
}

extern "C" void kernel_launch(void* const* d_in, const int* in_sizes, int n_in,
                              void* d_out, int out_size, void* d_ws, size_t ws_size,
                              hipStream_t stream) {
  const float* x     = (const float*)d_in[0];
  const float* wq    = (const float*)d_in[1];
  const float* wk    = (const float*)d_in[2];
  const float* wv    = (const float*)d_in[3];
  const float* wo    = (const float*)d_in[4];
  const float* gamma = (const float*)d_in[5];
  float* out        = (float*)d_out;
  float* weightsOut = out + OUT_ELEMS;

  // workspace layout (~37.2 MiB total)
  char* ws = (char*)d_ws;
  const size_t FRAGB = (size_t)BSZ * HEADS * EL_PER_BH * 2;  // 7,077,888 B each
  unsigned short* qf = (unsigned short*)(ws);
  unsigned short* kf = (unsigned short*)(ws + FRAGB);
  unsigned short* vf = (unsigned short*)(ws + 2 * FRAGB);
  float* Obuf  = (float*)(ws + 3 * FRAGB);                       // B*N*HK f32
  float* Ybuf  = (float*)(ws + 3 * FRAGB + (size_t)14155776);    // B*C*N f32
  float* gstat = (float*)(ws + 3 * FRAGB + (size_t)14155776 + (size_t)3538944);

  zero_stats_kernel<<<1, 256, 0, stream>>>(gstat);
  proj_frag_kernel<<<13824, 256, 0, stream>>>(x, wq, qf, 0);
  proj_frag_kernel<<<13824, 256, 0, stream>>>(x, wk, kf, 1);
  proj_frag_kernel<<<13824, 256, 0, stream>>>(x, wv, vf, 2);
  attn_kernel<<<BSZ * HEADS * NT16, 128, 0, stream>>>(qf, kf, vf, weightsOut, Obuf);
  outconv_kernel<<<BSZ * 432, 256, 0, stream>>>(Obuf, wo, Ybuf, gstat, gstat + 512);
  norm_kernel<<<BSZ * CCH, 256, 0, stream>>>(Ybuf, x, gstat, gstat + 512, gamma, out);
}
